// RecurrentVariationalPredictor_56848187130510
// MI455X (gfx1250) — matrix operand order repaired and checked by reading, not compiled, not run on hardware
//
#include <hip/hip_runtime.h>
#include <hip/hip_bf16.h>

// ---------------------------------------------------------------------------
// RecurrentVariationalPredictor on gfx1250 (CDNA5, wave32, WMMA)
// B=256, S=512, IN=64, H=512, L=64, O=256
// Persistent-kernel design: one launch runs all 512 timesteps with a
// device-wide generation barrier; cell state c lives in registers; the
// workgroup-shared A tiles (h_{t-1}, x_t) are staged into LDS each step with
// GLOBAL_LOAD_ASYNC_TO_LDS_B128 (ASYNCcnt) to kill 8x-redundant L2 fetches.
// ---------------------------------------------------------------------------

typedef __attribute__((ext_vector_type(16))) _Float16 v16h;
typedef __attribute__((ext_vector_type(8)))  _Float16 v8h;
typedef __attribute__((ext_vector_type(8)))  float    v8f;

constexpr int Bk  = 256;
constexpr int Sk  = 512;
constexpr int INk = 64;
constexpr int Hk  = 512;
constexpr int Lk  = 64;
constexpr int Ok  = 256;
constexpr int PKk = Hk + Lk;   // 576, W_pred row stride
constexpr int NWG = 64;        // persistent grid: 64 WGs x 8 waves = 512 waves

// ---- WMMA fragment helpers (layouts per CDNA5 ISA 7.12.2) -----------------
// A fragment: 16x32 f16 (MxK), row-major source, stride ld.
// lane l: M=l&15; halves[0..7]=K=8*(l>>4)+0..7 ; halves[8..15]=K=16+8*(l>>4)+0..7
__device__ inline v16h load_frag_a(const _Float16* __restrict__ p, int ld, int lane) {
  const int m = lane & 15;
  const int g = lane >> 4;
  const _Float16* row = p + m * ld;
  v8h lo = *(const v8h*)(row + 8 * g);
  v8h hi = *(const v8h*)(row + 16 + 8 * g);
  return __builtin_shufflevector(lo, hi, 0,1,2,3,4,5,6,7,8,9,10,11,12,13,14,15);
}

// B fragment: 32x16 (KxN), B = W^T with W row-major [N x K], stride ld.
// lane l: N=l&15; halves[0..7]=K=8*(l>>4)+0..7 ; halves[8..15]=K=16+8*(l>>4)+0..7
__device__ inline v16h load_frag_b(const _Float16* __restrict__ p, int ld, int lane) {
  const int n = lane & 15;
  const int g = lane >> 4;
  const _Float16* row = p + n * ld;
  v8h lo = *(const v8h*)(row + 8 * g);
  v8h hi = *(const v8h*)(row + 16 + 8 * g);
  return __builtin_shufflevector(lo, hi, 0,1,2,3,4,5,6,7,8,9,10,11,12,13,14,15);
}

__device__ inline v8f wmma_f16(v16h a, v16h b, v8f c) {
  return __builtin_amdgcn_wmma_f32_16x16x32_f16(false, a, false, b, (short)0, c, false, false);
}

__device__ inline float sigmoidf_(float x) { return 1.f / (1.f + __expf(-x)); }

// ---- async global -> LDS copy (16 bytes per active lane) -------------------
// ISA 15.18.3 op 98: GLOBAL_LOAD_ASYNC_TO_LDS_B128 (tracked with ASYNCcnt).
__device__ __forceinline__ void async_copy_b128(unsigned lds_off, const void* gaddr) {
  asm volatile("global_load_async_to_lds_b128 %0, %1, off"
               :: "v"(lds_off), "v"(gaddr) : "memory");
}
__device__ __forceinline__ void wait_asynccnt0() {
  asm volatile("s_wait_asynccnt 0x0" ::: "memory");
}

// ---- device-wide generation barrier ---------------------------------------
__device__ __forceinline__ void grid_barrier(unsigned* cnt, unsigned* gen) {
  __syncthreads();
  if (threadIdx.x == 0) {
    unsigned g = __hip_atomic_load(gen, __ATOMIC_RELAXED, __HIP_MEMORY_SCOPE_AGENT);
    unsigned arrived =
        __hip_atomic_fetch_add(cnt, 1u, __ATOMIC_ACQ_REL, __HIP_MEMORY_SCOPE_AGENT);
    if (arrived == (unsigned)(NWG - 1)) {
      __hip_atomic_store(cnt, 0u, __ATOMIC_RELAXED, __HIP_MEMORY_SCOPE_AGENT);
      __hip_atomic_fetch_add(gen, 1u, __ATOMIC_RELEASE, __HIP_MEMORY_SCOPE_AGENT);
    } else {
      while (__hip_atomic_load(gen, __ATOMIC_ACQUIRE, __HIP_MEMORY_SCOPE_AGENT) == g) {
        __builtin_amdgcn_s_sleep(2);
      }
    }
  }
  __syncthreads();
}

// ---- conversion / init kernels --------------------------------------------
__global__ void f32_to_f16_kernel(const float* __restrict__ src,
                                  _Float16* __restrict__ dst, int n) {
  int i = blockIdx.x * blockDim.x + threadIdx.x;
  if (i < n) dst[i] = (_Float16)src[i];
}

// x_seq [B,S,IN] f32 -> xh [S,B,IN] f16 (time-major for contiguous A tiles)
__global__ void convert_x_kernel(const float* __restrict__ x,
                                 _Float16* __restrict__ xh) {
  int i = blockIdx.x * blockDim.x + threadIdx.x;  // i = (s*256 + b)*64 + k
  int k = i & 63;
  int b = (i >> 6) & 255;
  int s = i >> 14;
  xh[i] = (_Float16)x[((size_t)b * Sk + s) * INk + k];
}

__global__ void zero_f32_kernel(float* __restrict__ p, int n) {
  int i = blockIdx.x * blockDim.x + threadIdx.x;
  if (i < n) p[i] = 0.f;
}

// ---------------------------------------------------------------------------
// Persistent LSTM+VAE kernel. 64 WGs x 256 threads (512 waves).
// Phase A (all 512 waves): wave (m_tile = gw>>5 [WG-uniform], n_tile = gw&31)
//   computes one 16x16 h tile: 4 gate accumulators over K=64 (x_t, LDS) +
//   K=512 (h_{t-1}, LDS), then the cell update with c kept in registers.
// bar -> Phase B1: waves 0..63 compute mu+lv tile pair (shared A frag) and z;
//   waves 64..319 accumulate the h-part of one 16x16 logits tile.
// bar -> Phase B2: logits waves add the z-part (K=64) and store.
// ---------------------------------------------------------------------------
__global__ __launch_bounds__(256) void lstm_persistent_kernel(
    const _Float16* __restrict__ xh,   // [S,B,IN]
    const _Float16* __restrict__ Wih,  // [4H,IN]
    const _Float16* __restrict__ Whh,  // [4H,H]
    const _Float16* __restrict__ Wmu,  // [L,H]
    const _Float16* __restrict__ Wlv,  // [L,H]
    const _Float16* __restrict__ Wpr,  // [O,H+L]
    const float* __restrict__ b_ih, const float* __restrict__ b_hh,
    const float* __restrict__ b_mu, const float* __restrict__ b_lv,
    const float* __restrict__ b_pred,
    const float* __restrict__ eps,     // [B,S,L]
    _Float16* __restrict__ hbuf0, _Float16* __restrict__ hbuf1,  // [B,H] x2
    _Float16* __restrict__ zbuf0, _Float16* __restrict__ zbuf1,  // [B,L] x2
    unsigned* __restrict__ bar_cnt, unsigned* __restrict__ bar_gen,
    float* __restrict__ out_logits, float* __restrict__ out_mu,
    float* __restrict__ out_lv, float* __restrict__ out_z) {
  __shared__ __align__(16) _Float16 sAh[16 * Hk];   // 16 KB: h_{t-1} rows of this WG
  __shared__ __align__(16) _Float16 sAx[16 * INk];  //  2 KB: x_t rows of this WG

  const int tid  = threadIdx.x;
  const int lane = tid & 31;
  const int wave = tid >> 5;
  const int gw   = blockIdx.x * 8 + wave;

  const unsigned ldsh = (unsigned)(uintptr_t)(void*)sAh;
  const unsigned ldsx = (unsigned)(uintptr_t)(void*)sAx;

  // ---- phase A tile + hoisted biases ----
  const int a_n0  = (gw & 31) * 16;
  const int wg_m0 = (blockIdx.x >> 2) * 16;  // == (gw>>5)*16, uniform per WG
  const int an    = a_n0 + (lane & 15);
  const float bi = b_ih[0 * Hk + an] + b_hh[0 * Hk + an];
  const float bf = b_ih[1 * Hk + an] + b_hh[1 * Hk + an];
  const float bg = b_ih[2 * Hk + an] + b_hh[2 * Hk + an];
  const float bo = b_ih[3 * Hk + an] + b_hh[3 * Hk + an];
  float c_reg[8] = {0.f, 0.f, 0.f, 0.f, 0.f, 0.f, 0.f, 0.f};

  // ---- phase B roles (wave-uniform) ----
  const bool do_mulv = (gw < 64);
  const int  b1_m0 = (gw >> 2) * 16;
  const int  b1_l  = (gw & 3) * 16 + (lane & 15);
  const float bmu = b_mu[b1_l & 63];
  const float blv = b_lv[b1_l & 63];

  const bool do_logits = (gw >= 64) && (gw < 320);
  const int  q      = gw - 64;
  const int  lg_m0  = (q >> 4) * 16;
  const int  lg_n0  = (q & 15) * 16;
  const float bpr   = b_pred[(lg_n0 + (lane & 15)) & (Ok - 1)];

  const v8f vzero = {};

  for (int t = 0; t < Sk; ++t) {
    const _Float16* hprev = (t & 1) ? hbuf1 : hbuf0;
    _Float16* hnext       = (t & 1) ? hbuf0 : hbuf1;
    _Float16* zb          = (t & 1) ? zbuf0 : zbuf1;  // parity of hnext

    // ---- stage this WG's contiguous A tiles into LDS (async DMA) ----
    {
      const char* hsrc = (const char*)(hprev + (size_t)wg_m0 * Hk);  // 16 KB
#pragma unroll
      for (int i = 0; i < 4; ++i) {
        async_copy_b128(ldsh + (unsigned)(tid * 16 + i * 4096),
                        hsrc + tid * 16 + i * 4096);
      }
      const char* xsrc = (const char*)(xh + ((size_t)t * Bk + wg_m0) * INk);  // 2 KB
      if (tid < 128) async_copy_b128(ldsx + (unsigned)(tid * 16), xsrc + tid * 16);
      wait_asynccnt0();
      __syncthreads();
    }

    // ================= phase A: gates + cell update =================
    {
      v8f acc[4] = {vzero, vzero, vzero, vzero};

#pragma unroll
      for (int kk = 0; kk < INk; kk += 32) {
        v16h a = load_frag_a(sAx + kk, INk, lane);
#pragma unroll
        for (int g = 0; g < 4; ++g) {
          v16h b = load_frag_b(Wih + (size_t)(g * Hk + a_n0) * INk + kk, INk, lane);
          acc[g] = wmma_f16(a, b, acc[g]);
        }
      }
#pragma unroll 2
      for (int kk = 0; kk < Hk; kk += 32) {
        v16h a = load_frag_a(sAh + kk, Hk, lane);
#pragma unroll
        for (int g = 0; g < 4; ++g) {
          v16h b = load_frag_b(Whh + (size_t)(g * Hk + a_n0) * Hk + kk, Hk, lane);
          acc[g] = wmma_f16(a, b, acc[g]);
        }
      }
#pragma unroll
      for (int r = 0; r < 8; ++r) {
        const float ig = sigmoidf_(acc[0][r] + bi);
        const float fg = sigmoidf_(acc[1][r] + bf);
        const float gg = tanhf(acc[2][r] + bg);
        const float og = sigmoidf_(acc[3][r] + bo);
        c_reg[r] = fg * c_reg[r] + ig * gg;
        const int m = wg_m0 + r + 8 * (lane >> 4);
        hnext[(size_t)m * Hk + an] = (_Float16)(og * tanhf(c_reg[r]));
      }
    }

    grid_barrier(bar_cnt, bar_gen);  // h(t) visible everywhere

    // ================= phase B1 =================
    v8f accL = vzero;  // logits accumulator persists across the barrier
    if (do_mulv) {
      v8f am = vzero, al = vzero;
      const _Float16* hrow = hnext + (size_t)b1_m0 * Hk;
      const int j16 = (b1_l & 48);  // j*16 (row block of Wmu/Wlv)
#pragma unroll 2
      for (int kk = 0; kk < Hk; kk += 32) {
        v16h a  = load_frag_a(hrow + kk, Hk, lane);
        v16h bm = load_frag_b(Wmu + (size_t)j16 * Hk + kk, Hk, lane);
        v16h bl = load_frag_b(Wlv + (size_t)j16 * Hk + kk, Hk, lane);
        am = wmma_f16(a, bm, am);
        al = wmma_f16(a, bl, al);
      }
#pragma unroll
      for (int r = 0; r < 8; ++r) {
        const int m = b1_m0 + r + 8 * (lane >> 4);
        const float mu = am[r] + bmu;
        const float lv = al[r] + blv;
        const size_t oidx = ((size_t)m * Sk + t) * Lk + b1_l;
        out_mu[oidx] = mu;
        out_lv[oidx] = lv;
        const float z = mu + eps[oidx] * __expf(0.5f * lv);
        out_z[oidx] = z;
        zb[(size_t)m * Lk + b1_l] = (_Float16)z;
      }
    } else if (do_logits) {
      const _Float16* hrow = hnext + (size_t)lg_m0 * Hk;
#pragma unroll 2
      for (int kk = 0; kk < Hk; kk += 32) {
        v16h a = load_frag_a(hrow + kk, Hk, lane);
        v16h b = load_frag_b(Wpr + (size_t)lg_n0 * PKk + kk, PKk, lane);
        accL = wmma_f16(a, b, accL);
      }
    }

    grid_barrier(bar_cnt, bar_gen);  // z(t) visible everywhere

    // ================= phase B2: logits z-part + store =================
    if (do_logits) {
#pragma unroll
      for (int kk = 0; kk < Lk; kk += 32) {
        v16h a = load_frag_a(zb + (size_t)lg_m0 * Lk + kk, Lk, lane);
        v16h b = load_frag_b(Wpr + (size_t)lg_n0 * PKk + Hk + kk, PKk, lane);
        accL = wmma_f16(a, b, accL);
      }
      const int nn = lg_n0 + (lane & 15);
#pragma unroll
      for (int r = 0; r < 8; ++r) {
        const int m = lg_m0 + r + 8 * (lane >> 4);
        out_logits[((size_t)m * Sk + t) * Ok + nn] = accL[r] + bpr;
      }
    }
    // No barrier needed before A(t+1): it writes the other h buffer and
    // touches only wave-private c; B2 readers use hnext/zb of this step;
    // the LDS restage is protected by the two grid barriers in between.
  }
}

// ---------------------------------------------------------------------------
extern "C" void kernel_launch(void* const* d_in, const int* in_sizes, int n_in,
                              void* d_out, int out_size, void* d_ws, size_t ws_size,
                              hipStream_t stream) {
  (void)in_sizes; (void)n_in; (void)out_size; (void)ws_size;

  const float* x_seq  = (const float*)d_in[0];
  const float* eps    = (const float*)d_in[1];
  const float* W_ih   = (const float*)d_in[2];
  const float* b_ih   = (const float*)d_in[3];
  const float* W_hh   = (const float*)d_in[4];
  const float* b_hh   = (const float*)d_in[5];
  const float* W_mu   = (const float*)d_in[6];
  const float* b_mu   = (const float*)d_in[7];
  const float* W_lv   = (const float*)d_in[8];
  const float* b_lv   = (const float*)d_in[9];
  const float* W_pred = (const float*)d_in[10];
  const float* b_pred = (const float*)d_in[11];

  // workspace layout (bytes)
  char* ws = (char*)d_ws;
  _Float16* Wih_h = (_Float16*)(ws + 0);         //  262144 B (2048*64 f16)
  _Float16* Whh_h = (_Float16*)(ws + 262144);    // 2097152 B (2048*512 f16)
  _Float16* Wmu_h = (_Float16*)(ws + 2359296);   //   65536 B (64*512 f16)
  _Float16* Wlv_h = (_Float16*)(ws + 2424832);   //   65536 B
  _Float16* Wpr_h = (_Float16*)(ws + 2490368);   //  294912 B (256*576 f16)
  _Float16* xh    = (_Float16*)(ws + 2785280);   // 16777216 B (512*256*64 f16)
  _Float16* h0buf = (_Float16*)(ws + 19562496);  //  262144 B (256*512 f16)
  unsigned* barv  = (unsigned*)(ws + 19824640);  //     256 B (cnt, gen, pad)
  _Float16* h1buf = (_Float16*)(ws + 19824896);  //  262144 B
  _Float16* z0buf = (_Float16*)(ws + 20087040);  //   32768 B (256*64 f16)
  _Float16* z1buf = (_Float16*)(ws + 20119808);  //   32768 B  (total ~20.2 MB)

  float* out_logits = (float*)d_out;
  float* out_mu = out_logits + (size_t)Bk * Sk * Ok;
  float* out_lv = out_mu + (size_t)Bk * Sk * Lk;
  float* out_z  = out_lv + (size_t)Bk * Sk * Lk;

  // prep (re-done every call; deterministic)
  f32_to_f16_kernel<<<(131072 + 255) / 256, 256, 0, stream>>>(W_ih, Wih_h, 131072);
  f32_to_f16_kernel<<<(1048576 + 255) / 256, 256, 0, stream>>>(W_hh, Whh_h, 1048576);
  f32_to_f16_kernel<<<(32768 + 255) / 256, 256, 0, stream>>>(W_mu, Wmu_h, 32768);
  f32_to_f16_kernel<<<(32768 + 255) / 256, 256, 0, stream>>>(W_lv, Wlv_h, 32768);
  f32_to_f16_kernel<<<(147456 + 255) / 256, 256, 0, stream>>>(W_pred, Wpr_h, 147456);
  convert_x_kernel<<<(Bk * Sk * INk) / 256, 256, 0, stream>>>(x_seq, xh);
  // zero h0 + barrier vars (contiguous: 262144 + 256 bytes = 65600 f32)
  zero_f32_kernel<<<(65600 + 255) / 256, 256, 0, stream>>>((float*)(ws + 19562496), 65600);

  lstm_persistent_kernel<<<NWG, 256, 0, stream>>>(
      xh, Wih_h, Whh_h, Wmu_h, Wlv_h, Wpr_h,
      b_ih, b_hh, b_mu, b_lv, b_pred, eps,
      h0buf, h1buf, z0buf, z1buf,
      barv + 0, barv + 1,
      out_logits, out_mu, out_lv, out_z);
}
